// Attention_53712861004017
// MI455X (gfx1250) — compile-verified, matrix operand order checked by reading, not run on hardware
//
#include <hip/hip_runtime.h>
#include <hip/hip_bf16.h>
#include <math.h>

#define DIM   768
#define HEADS 12
#define HD    64
#define NSEQ  2048
#define NB    4
#define NCLS  1000
#define SCALE 0.125f

typedef __attribute__((ext_vector_type(16))) __bf16 v16bf;
typedef __attribute__((ext_vector_type(8)))  __bf16 v8bf;
typedef __attribute__((ext_vector_type(8)))  float  v8f;
typedef int v4i_vs __attribute__((vector_size(16)));   // matches builtin param type

__device__ __forceinline__ unsigned short f2bf(float f) {
  unsigned u = __float_as_uint(f);
  u += 0x7FFFu + ((u >> 16) & 1u);            // round-to-nearest-even
  return (unsigned short)(u >> 16);
}

// Build a v16bf A/B fragment from two contiguous 8-element (16B) chunks.
// Per ISA layout: VGPR0-3 <- first 16B, VGPR4-7 <- second 16B.
__device__ __forceinline__ v16bf ldfrag(const unsigned short* p0, const unsigned short* p1) {
  union { v16bf v; v8bf h[2]; } u;
  u.h[0] = *(const v8bf*)p0;
  u.h[1] = *(const v8bf*)p1;
  return u.v;
}

__device__ __forceinline__ v8f wmma_bf16(v16bf a, v16bf b, v8f c) {
  return __builtin_amdgcn_wmma_f32_16x16x32_bf16(false, a, false, b, (short)0, c, false, false);
}

// gfx1250 async direct memory->LDS copy (16B per lane), tracked by ASYNCcnt.
__device__ __forceinline__ void async_cp16(void* lds, const void* g) {
  __builtin_amdgcn_global_load_async_to_lds_b128(
      (__attribute__((address_space(1))) v4i_vs*)(g),
      (__attribute__((address_space(3))) v4i_vs*)(lds), 0, 0);
}
__device__ __forceinline__ void wait_async0() {
  asm volatile("s_wait_asynccnt 0x0" ::: "memory");
}

// Monotone float<->u32 encoding so max() can use atomicMax(u32).
__device__ __forceinline__ unsigned fenc(float f) {
  unsigned u = __float_as_uint(f);
  return (u & 0x80000000u) ? ~u : (u | 0x80000000u);
}
__device__ __forceinline__ float fdec(unsigned e) {
  unsigned u = (e & 0x80000000u) ? (e & 0x7FFFFFFFu) : ~e;
  return __uint_as_float(u);
}

// ---------------------------------------------------------------------------
// Kernel 1: qkv = x @ w_qkv  (fp32 in, bf16 out, split into q/k/v [B,H,N,64])
// Block tile 64x64, K-step 32, 8 waves as 4(M)x2(N), wave tile 16x32.
// Software pipelined: next K-tile loaded into registers during WMMA phase.
// ---------------------------------------------------------------------------
__global__ __launch_bounds__(256) void k_qkv(const float* __restrict__ x,
                                             const float* __restrict__ w,
                                             unsigned short* __restrict__ qb,
                                             unsigned short* __restrict__ kb,
                                             unsigned short* __restrict__ vb) {
  __shared__ unsigned short As[64][40];   // [row][k], stride 40 (80B, 16B-mult)
  __shared__ unsigned short Bs[64][40];   // [col][k] (w transposed)
  const int n0 = blockIdx.x * 64;
  const int m0 = blockIdx.y * 64;
  const int tid = threadIdx.x;
  const int wave = tid >> 5, lane = tid & 31;
  const int half = lane >> 4, ln = lane & 15;
  const int wm = wave >> 1, wn = wave & 1;
  const int k0 = half ? 8 : 0;

  // Staging slot coordinates (2 x float4 each for A and B per thread).
  const int a0r = tid >> 3,       ac = (tid & 7) << 2;   // A: 64x32
  const int a1r = a0r + 32;
  const int b0r = tid >> 4,       bc = (tid & 15) << 2;  // B: 32x64
  const int b1r = b0r + 16;

  const v8f vzero = {0.f,0.f,0.f,0.f,0.f,0.f,0.f,0.f};
  v8f acc[2]; acc[0] = vzero; acc[1] = vzero;

  float4 ra0, ra1, rb0, rb1;
  ra0 = *(const float4*)(x + (size_t)(m0 + a0r) * DIM + ac);
  ra1 = *(const float4*)(x + (size_t)(m0 + a1r) * DIM + ac);
  rb0 = *(const float4*)(w + (size_t)b0r * (3 * DIM) + n0 + bc);
  rb1 = *(const float4*)(w + (size_t)b1r * (3 * DIM) + n0 + bc);

  for (int kt = 0; kt < DIM; kt += 32) {
    if (kt) __syncthreads();
    As[a0r][ac]   = f2bf(ra0.x); As[a0r][ac+1] = f2bf(ra0.y);
    As[a0r][ac+2] = f2bf(ra0.z); As[a0r][ac+3] = f2bf(ra0.w);
    As[a1r][ac]   = f2bf(ra1.x); As[a1r][ac+1] = f2bf(ra1.y);
    As[a1r][ac+2] = f2bf(ra1.z); As[a1r][ac+3] = f2bf(ra1.w);
    Bs[bc][b0r]   = f2bf(rb0.x); Bs[bc+1][b0r] = f2bf(rb0.y);
    Bs[bc+2][b0r] = f2bf(rb0.z); Bs[bc+3][b0r] = f2bf(rb0.w);
    Bs[bc][b1r]   = f2bf(rb1.x); Bs[bc+1][b1r] = f2bf(rb1.y);
    Bs[bc+2][b1r] = f2bf(rb1.z); Bs[bc+3][b1r] = f2bf(rb1.w);
    __syncthreads();
    const int kn = kt + 32;
    if (kn < DIM) {                      // prefetch next tile (overlaps WMMA)
      ra0 = *(const float4*)(x + (size_t)(m0 + a0r) * DIM + kn + ac);
      ra1 = *(const float4*)(x + (size_t)(m0 + a1r) * DIM + kn + ac);
      rb0 = *(const float4*)(w + (size_t)(kn + b0r) * (3 * DIM) + n0 + bc);
      rb1 = *(const float4*)(w + (size_t)(kn + b1r) * (3 * DIM) + n0 + bc);
    }
    const unsigned short* ar = &As[wm * 16 + ln][0];
    const v16bf a = ldfrag(ar + k0, ar + k0 + 16);
#pragma unroll
    for (int t = 0; t < 2; ++t) {
      const unsigned short* br = &Bs[wn * 32 + t * 16 + ln][0];
      const v16bf bf = ldfrag(br + k0, br + k0 + 16);
      acc[t] = wmma_bf16(a, bf, acc[t]);
    }
  }
  // Epilogue: scatter to q/k/v in [B,H,N,64] bf16.
#pragma unroll
  for (int t = 0; t < 2; ++t) {
    const int col = n0 + wn * 32 + t * 16 + ln;
    const int which = col / DIM, rem = col % DIM;
    const int h = rem >> 6, d = rem & 63;
    unsigned short* dst = (which == 0) ? qb : ((which == 1) ? kb : vb);
#pragma unroll
    for (int r = 0; r < 8; ++r) {
      const int row = m0 + wm * 16 + r + half * 8;
      const int bidx = row >> 11, nidx = row & (NSEQ - 1);
      dst[(((size_t)bidx * HEADS + h) * NSEQ + nidx) * HD + d] = f2bf(acc[t][r]);
    }
  }
}

// ---------------------------------------------------------------------------
// Kernel 2: flash attention. One block = one (b,h) x 128 query rows.
// K tile staged via async global->LDS; V tile register-prefetched + transposed.
// ---------------------------------------------------------------------------
__global__ __launch_bounds__(256) void k_attn(const unsigned short* __restrict__ qb,
                                              const unsigned short* __restrict__ kb,
                                              const unsigned short* __restrict__ vb,
                                              unsigned short* __restrict__ ao) {
  __shared__ unsigned short Ks[64][72];      // [key][d]
  __shared__ unsigned short Vt[64][72];      // [d][key]  (V transposed)
  __shared__ unsigned short Ps[8][16][72];   // per-wave P tile [row][key]
  const int bh = blockIdx.y;
  const int b = bh / HEADS, h = bh % HEADS;
  const int r0 = blockIdx.x * 128;
  const int tid = threadIdx.x;
  const int wave = tid >> 5, lane = tid & 31;
  const int half = lane >> 4, ln = lane & 15;
  const int k0 = half ? 8 : 0;

  const size_t headbase = (size_t)bh * NSEQ * HD;
  const unsigned short* qp = qb + headbase + (size_t)(r0 + wave * 16 + ln) * HD;
  const v16bf qa0 = ldfrag(qp + k0, qp + k0 + 16);           // d 0..31
  const v16bf qa1 = ldfrag(qp + 32 + k0, qp + 32 + k0 + 16); // d 32..63

  // Staging slots: 2 x 16B per thread over a 64x64 bf16 tile.
  const int sr0 = tid >> 3, sc = (tid & 7) << 3;
  const int sr1 = sr0 + 32;

  const v8f vzero = {0.f,0.f,0.f,0.f,0.f,0.f,0.f,0.f};
  float m[8], l[8];
  v8f o[4];
#pragma unroll
  for (int r = 0; r < 8; ++r) { m[r] = -1e30f; l[r] = 0.f; }
#pragma unroll
  for (int t = 0; t < 4; ++t) o[t] = vzero;

  union V8 { uint4 q; unsigned short s16[8]; };
  V8 rv0, rv1;                               // V prefetch registers
  rv0.q = *(const uint4*)(vb + headbase + (size_t)sr0 * HD + sc);
  rv1.q = *(const uint4*)(vb + headbase + (size_t)sr1 * HD + sc);

  for (int it = 0; it < NSEQ / 64; ++it) {
    const int c0 = it * 64;
    __syncthreads();
    // K tile: direct memory->LDS async copy (no VGPR round-trip).
    async_cp16(&Ks[sr0][sc], kb + headbase + (size_t)(c0 + sr0) * HD + sc);
    async_cp16(&Ks[sr1][sc], kb + headbase + (size_t)(c0 + sr1) * HD + sc);
    // V tile: transpose-store the prefetched registers.
#pragma unroll
    for (int i = 0; i < 8; ++i) Vt[sc + i][sr0] = rv0.s16[i];
#pragma unroll
    for (int i = 0; i < 8; ++i) Vt[sc + i][sr1] = rv1.s16[i];
    wait_async0();
    __syncthreads();
    if (it + 1 < NSEQ / 64) {                // prefetch next V (overlaps WMMA)
      const size_t gn = headbase + (size_t)(c0 + 64) * HD + sc;
      rv0.q = *(const uint4*)(vb + gn + (size_t)sr0 * HD);
      rv1.q = *(const uint4*)(vb + gn + (size_t)sr1 * HD);
    }

    v8f sv[4];                                // S = Q @ K^T  (4 key sub-tiles)
#pragma unroll
    for (int t = 0; t < 4; ++t) {
      const unsigned short* kr = &Ks[t * 16 + ln][0];
      const v16bf b0 = ldfrag(kr + k0, kr + k0 + 16);
      const v16bf b1 = ldfrag(kr + 32 + k0, kr + 32 + k0 + 16);
      v8f a = vzero;
      a = wmma_bf16(qa0, b0, a);
      a = wmma_bf16(qa1, b1, a);
      sv[t] = a;
    }
    float al[8];                              // online softmax
#pragma unroll
    for (int r = 0; r < 8; ++r) {
      float v = fmaxf(fmaxf(sv[0][r], sv[1][r]), fmaxf(sv[2][r], sv[3][r])) * SCALE;
      v = fmaxf(v, __shfl_xor(v, 1, 16));
      v = fmaxf(v, __shfl_xor(v, 2, 16));
      v = fmaxf(v, __shfl_xor(v, 4, 16));
      v = fmaxf(v, __shfl_xor(v, 8, 16));
      const float mn = fmaxf(m[r], v);
      al[r] = __expf(m[r] - mn);
      m[r] = mn;
    }
#pragma unroll
    for (int t = 0; t < 4; ++t)
#pragma unroll
      for (int r = 0; r < 8; ++r)
        sv[t][r] = __expf(sv[t][r] * SCALE - m[r]);
#pragma unroll
    for (int r = 0; r < 8; ++r) {
      float rs = sv[0][r] + sv[1][r] + sv[2][r] + sv[3][r];
      rs += __shfl_xor(rs, 1, 16);
      rs += __shfl_xor(rs, 2, 16);
      rs += __shfl_xor(rs, 4, 16);
      rs += __shfl_xor(rs, 8, 16);
      l[r] = l[r] * al[r] + rs;
    }
#pragma unroll
    for (int t = 0; t < 4; ++t)               // P -> per-wave LDS (C/D->A transpose)
#pragma unroll
      for (int r = 0; r < 8; ++r)
        Ps[wave][r + half * 8][t * 16 + ln] = f2bf(sv[t][r]);
#pragma unroll
    for (int t = 0; t < 4; ++t)               // rescale O
#pragma unroll
      for (int r = 0; r < 8; ++r)
        o[t][r] *= al[r];
    const unsigned short* pr = &Ps[wave][ln][0];
    const v16bf pa0 = ldfrag(pr + k0, pr + k0 + 16);
    const v16bf pa1 = ldfrag(pr + 32 + k0, pr + 32 + k0 + 16);
#pragma unroll
    for (int t = 0; t < 4; ++t) {             // O += P @ V
      const unsigned short* vr = &Vt[t * 16 + ln][0];
      const v16bf b0 = ldfrag(vr + k0, vr + k0 + 16);
      const v16bf b1 = ldfrag(vr + 32 + k0, vr + 32 + k0 + 16);
      o[t] = wmma_bf16(pa0, b0, o[t]);
      o[t] = wmma_bf16(pa1, b1, o[t]);
    }
  }
#pragma unroll
  for (int r = 0; r < 8; ++r) {               // finalize + store bf16 [B,N,768]
    const float inv = 1.f / l[r];
    const int row = r0 + wave * 16 + r + half * 8;
    unsigned short* op = ao + ((size_t)b * NSEQ + row) * DIM + h * HD;
#pragma unroll
    for (int t = 0; t < 4; ++t)
      op[t * 16 + ln] = f2bf(o[t][r] * inv);
  }
}

// ---------------------------------------------------------------------------
// Kernel 3: y1 = attn_out @ w_proj + b_proj.  bf16 A tile staged via async
// global->LDS; fp32 W tile register-prefetched and converted.
// ---------------------------------------------------------------------------
__global__ __launch_bounds__(256) void k_proj(const unsigned short* __restrict__ ao,
                                              const float* __restrict__ w,
                                              const float* __restrict__ bias,
                                              unsigned short* __restrict__ y1) {
  __shared__ unsigned short As[64][40];
  __shared__ unsigned short Bs[64][40];
  const int n0 = blockIdx.x * 64;
  const int m0 = blockIdx.y * 64;
  const int tid = threadIdx.x;
  const int wave = tid >> 5, lane = tid & 31;
  const int half = lane >> 4, ln = lane & 15;
  const int wm = wave >> 1, wn = wave & 1;
  const int k0 = half ? 8 : 0;

  const int ar0 = tid >> 2, ac = (tid & 3) << 3;         // A: 64x32 bf16, 16B/thread
  const int b0r = tid >> 4, bc = (tid & 15) << 2;        // B: 32x64 f32
  const int b1r = b0r + 16;

  const v8f vzero = {0.f,0.f,0.f,0.f,0.f,0.f,0.f,0.f};
  v8f acc[2]; acc[0] = vzero; acc[1] = vzero;

  float4 rb0 = *(const float4*)(w + (size_t)b0r * DIM + n0 + bc);
  float4 rb1 = *(const float4*)(w + (size_t)b1r * DIM + n0 + bc);

  for (int kt = 0; kt < DIM; kt += 32) {
    if (kt) __syncthreads();
    async_cp16(&As[ar0][ac], ao + (size_t)(m0 + ar0) * DIM + kt + ac);
    Bs[bc][b0r]   = f2bf(rb0.x); Bs[bc+1][b0r] = f2bf(rb0.y);
    Bs[bc+2][b0r] = f2bf(rb0.z); Bs[bc+3][b0r] = f2bf(rb0.w);
    Bs[bc][b1r]   = f2bf(rb1.x); Bs[bc+1][b1r] = f2bf(rb1.y);
    Bs[bc+2][b1r] = f2bf(rb1.z); Bs[bc+3][b1r] = f2bf(rb1.w);
    wait_async0();
    __syncthreads();
    const int kn = kt + 32;
    if (kn < DIM) {
      rb0 = *(const float4*)(w + (size_t)(kn + b0r) * DIM + n0 + bc);
      rb1 = *(const float4*)(w + (size_t)(kn + b1r) * DIM + n0 + bc);
    }
    const unsigned short* arp = &As[wm * 16 + ln][0];
    const v16bf a = ldfrag(arp + k0, arp + k0 + 16);
#pragma unroll
    for (int t = 0; t < 2; ++t) {
      const unsigned short* br = &Bs[wn * 32 + t * 16 + ln][0];
      const v16bf bf = ldfrag(br + k0, br + k0 + 16);
      acc[t] = wmma_bf16(a, bf, acc[t]);
    }
  }
#pragma unroll
  for (int t = 0; t < 2; ++t) {
    const int col = n0 + wn * 32 + t * 16 + ln;
    const float bv = bias[col];
#pragma unroll
    for (int r = 0; r < 8; ++r) {
      const int row = m0 + wm * 16 + r + half * 8;
      y1[(size_t)row * DIM + col] = f2bf(acc[t][r] + bv);
    }
  }
}

// ---------------------------------------------------------------------------
// Kernel 4: y2 = y1 @ w_head (+b_head after max); block-local row-max, then
// monotone-encoded atomicMax(u32) into d_out[b][col].
// ---------------------------------------------------------------------------
__global__ __launch_bounds__(256) void k_head(const unsigned short* __restrict__ y1,
                                              const float* __restrict__ w,
                                              const float* __restrict__ bias,
                                              unsigned* __restrict__ oenc) {
  __shared__ unsigned short As[64][40];
  __shared__ unsigned short Bs[64][40];
  __shared__ float red[64][8];
  const int n0 = blockIdx.x * 64;
  const int m0 = blockIdx.y * 64;
  const int tid = threadIdx.x;
  const int wave = tid >> 5, lane = tid & 31;
  const int half = lane >> 4, ln = lane & 15;
  const int wm = wave >> 1, wn = wave & 1;
  const int k0 = half ? 8 : 0;

  const int ar0 = tid >> 2, ac = (tid & 3) << 3;
  const int b0r = tid >> 4, bc = (tid & 15) << 2;
  const int b1r = b0r + 16;
  const int gcol = n0 + bc;
  const bool bok = gcol < NCLS;   // NCLS % 4 == 0 -> float4 fully in/out

  const v8f vzero = {0.f,0.f,0.f,0.f,0.f,0.f,0.f,0.f};
  v8f acc[2]; acc[0] = vzero; acc[1] = vzero;

  float4 rb0 = {0.f,0.f,0.f,0.f}, rb1 = {0.f,0.f,0.f,0.f};
  if (bok) {
    rb0 = *(const float4*)(w + (size_t)b0r * NCLS + gcol);
    rb1 = *(const float4*)(w + (size_t)b1r * NCLS + gcol);
  }

  for (int kt = 0; kt < DIM; kt += 32) {
    if (kt) __syncthreads();
    async_cp16(&As[ar0][ac], y1 + (size_t)(m0 + ar0) * DIM + kt + ac);
    Bs[bc][b0r]   = f2bf(rb0.x); Bs[bc+1][b0r] = f2bf(rb0.y);
    Bs[bc+2][b0r] = f2bf(rb0.z); Bs[bc+3][b0r] = f2bf(rb0.w);
    Bs[bc][b1r]   = f2bf(rb1.x); Bs[bc+1][b1r] = f2bf(rb1.y);
    Bs[bc+2][b1r] = f2bf(rb1.z); Bs[bc+3][b1r] = f2bf(rb1.w);
    wait_async0();
    __syncthreads();
    const int kn = kt + 32;
    if (kn < DIM && bok) {
      rb0 = *(const float4*)(w + (size_t)(kn + b0r) * NCLS + gcol);
      rb1 = *(const float4*)(w + (size_t)(kn + b1r) * NCLS + gcol);
    }
    const unsigned short* arp = &As[wm * 16 + ln][0];
    const v16bf a = ldfrag(arp + k0, arp + k0 + 16);
#pragma unroll
    for (int t = 0; t < 2; ++t) {
      const unsigned short* br = &Bs[wn * 32 + t * 16 + ln][0];
      const v16bf bf = ldfrag(br + k0, br + k0 + 16);
      acc[t] = wmma_bf16(a, bf, acc[t]);
    }
  }
  // Per-column max over the block's 64 rows.
#pragma unroll
  for (int t = 0; t < 2; ++t) {
    float lm = acc[t][0];
#pragma unroll
    for (int r = 1; r < 8; ++r) lm = fmaxf(lm, acc[t][r]);
    red[wn * 32 + t * 16 + ln][wm * 2 + half] = lm;
  }
  __syncthreads();
  if (tid < 64) {
    float v = red[tid][0];
#pragma unroll
    for (int i = 1; i < 8; ++i) v = fmaxf(v, red[tid][i]);
    const int col = n0 + tid;
    if (col < NCLS)
      atomicMax(&oenc[(size_t)(m0 >> 11) * NCLS + col], fenc(v + bias[col]));
  }
}

__global__ void k_init(unsigned* o, int n) {
  const int i = blockIdx.x * 256 + threadIdx.x;
  if (i < n) o[i] = 0x007FFFFFu;               // fenc(-inf)
}
__global__ void k_dec(unsigned* o, int n) {
  const int i = blockIdx.x * 256 + threadIdx.x;
  if (i < n) {
    const float f = fdec(o[i]);
    ((float*)o)[i] = f;
  }
}

extern "C" void kernel_launch(void* const* d_in, const int* in_sizes, int n_in,
                              void* d_out, int out_size, void* d_ws, size_t ws_size,
                              hipStream_t stream) {
  const float* x      = (const float*)d_in[0];
  const float* w_qkv  = (const float*)d_in[1];
  const float* w_proj = (const float*)d_in[2];
  const float* b_proj = (const float*)d_in[3];
  const float* w_head = (const float*)d_in[4];
  const float* b_head = (const float*)d_in[5];

  unsigned short* ws = (unsigned short*)d_ws;
  const size_t HBUF = (size_t)NB * HEADS * NSEQ * HD;    // per q/k/v buffer
  unsigned short* qb = ws;
  unsigned short* kb = qb + HBUF;
  unsigned short* vb = kb + HBUF;
  unsigned short* ao = vb + HBUF;                        // [B*N, 768] bf16
  unsigned short* y1 = ao + (size_t)NB * NSEQ * DIM;     // [B*N, 768] bf16

  unsigned* oenc = (unsigned*)d_out;
  const int on = NB * NCLS;

  k_init<<<(on + 255) / 256, 256, 0, stream>>>(oenc, on);
  k_qkv <<<dim3((3 * DIM) / 64, (NB * NSEQ) / 64), 256, 0, stream>>>(x, w_qkv, qb, kb, vb);
  k_attn<<<dim3(NSEQ / 128, NB * HEADS),           256, 0, stream>>>(qb, kb, vb, ao);
  k_proj<<<dim3(DIM / 64, (NB * NSEQ) / 64),       256, 0, stream>>>(ao, w_proj, b_proj, y1);
  k_head<<<dim3((NCLS + 63) / 64, (NB * NSEQ) / 64), 256, 0, stream>>>(y1, w_head, b_head, oenc);
  k_dec <<<(on + 255) / 256, 256, 0, stream>>>(oenc, on);

  (void)in_sizes; (void)n_in; (void)out_size; (void)ws_size;
}
